// TrackMPNN_64639257805380
// MI455X (gfx1250) — matrix operand order repaired and checked by reading, not compiled, not run on hardware
//
#include <hip/hip_runtime.h>

#define N_NODES 8192
#define ADJ_LD  8192

typedef __attribute__((ext_vector_type(16))) __bf16 v16bf;
typedef __attribute__((ext_vector_type(8)))  float  v8f;
typedef int v2i __attribute__((vector_size(8)));     // matches async builtin's b64 pointee

#if defined(__has_builtin)
#if __has_builtin(__builtin_amdgcn_global_load_async_to_lds_b64)
#define HAVE_ASYNC_B64 1
#endif
#if __has_builtin(__builtin_amdgcn_s_wait_asynccnt)
#define HAVE_WAIT_ASYNC 1
#endif
#endif

__device__ __forceinline__ unsigned short f2bf(float f) {
  unsigned u = __float_as_uint(f);
  u += 0x7FFFu + ((u >> 16) & 1u);      // round-to-nearest-even
  return (unsigned short)(u >> 16);
}
// Truncating pack of two f32 -> packed bf16 pair in ONE v_perm_b32.
// Exact for values that are bf16-representable (adjacency entries are 0/1).
__device__ __forceinline__ unsigned pack2bf_trunc(float lo, float hi) {
  return __builtin_amdgcn_perm(__float_as_uint(hi), __float_as_uint(lo),
                               0x07060302u);
}

// ---------------------------------------------------------------------------
// CDNA5 async global->LDS copy (ASYNCcnt-tracked) + wait.
// ---------------------------------------------------------------------------
__device__ __forceinline__ void async_copy_b64(const void* g, void* l) {
#if defined(HAVE_ASYNC_B64)
  __builtin_amdgcn_global_load_async_to_lds_b64(
      (__attribute__((address_space(1))) v2i*)g,
      (__attribute__((address_space(3))) v2i*)l, 0, 0);
#else
  unsigned lds = (unsigned)(unsigned long long)l;   // low 32 bits = LDS offset
  asm volatile("global_load_async_to_lds_b64 %0, %1, off"
               :: "v"(lds), "v"(g) : "memory");
#endif
}
__device__ __forceinline__ void wait_async0() {
#if defined(HAVE_WAIT_ASYNC)
  __builtin_amdgcn_s_wait_asynccnt(0);
#else
  asm volatile("s_wait_asynccnt 0x0" ::: "memory");
#endif
}

// ---------------------------------------------------------------------------
// PointNet feature extractor: per row, 3->64->128->64 1x1 convs over 5 points,
// max-pool, then h0 = concat(x[:, :64], feat).  Emits fp32 row-major h0 and a
// TRANSPOSED bf16 copy h0bT[c][row] (feeds the async B-tile path of the GEMM).
// ---------------------------------------------------------------------------
__global__ __launch_bounds__(64) void pointnet_kernel(
    const float* __restrict__ x,
    const float* __restrict__ w1, const float* __restrict__ b1,
    const float* __restrict__ w2, const float* __restrict__ b2,
    const float* __restrict__ w3, const float* __restrict__ b3,
    float* __restrict__ h0, unsigned short* __restrict__ h0bT) {
  __shared__ float h1s[64 * 5];
  __shared__ float h2s[128 * 5];
  const int row = blockIdx.x;
  const int t   = threadIdx.x;
  const float* xr = x + (size_t)row * 74;

  float c0[5], c1[5];
#pragma unroll
  for (int p = 0; p < 5; p++) { c0[p] = xr[64 + p]; c1[p] = xr[69 + p]; }

  {
    float w10 = w1[t * 3 + 0], w11 = w1[t * 3 + 1], bb = b1[t];
#pragma unroll
    for (int p = 0; p < 5; p++) {
      float v = fmaf(w10, c0[p], fmaf(w11, c1[p], bb));
      h1s[t * 5 + p] = v > 0.f ? v : 0.f;
    }
  }
  __syncthreads();

#pragma unroll
  for (int oo = 0; oo < 2; oo++) {
    int o = t + oo * 64;
    float acc[5];
    float bb = b2[o];
#pragma unroll
    for (int p = 0; p < 5; p++) acc[p] = bb;
    for (int c = 0; c < 64; c++) {
      float wv = w2[o * 64 + c];
#pragma unroll
      for (int p = 0; p < 5; p++) acc[p] = fmaf(wv, h1s[c * 5 + p], acc[p]);
    }
#pragma unroll
    for (int p = 0; p < 5; p++) h2s[o * 5 + p] = acc[p] > 0.f ? acc[p] : 0.f;
  }
  __syncthreads();

  float acc[5];
  float bb = b3[t];
#pragma unroll
  for (int p = 0; p < 5; p++) acc[p] = bb;
  for (int c = 0; c < 128; c++) {
    float wv = w3[t * 128 + c];
#pragma unroll
    for (int p = 0; p < 5; p++) acc[p] = fmaf(wv, h2s[c * 5 + p], acc[p]);
  }
  float m = acc[0];
#pragma unroll
  for (int p = 1; p < 5; p++) m = fmaxf(m, acc[p]);

  float xv = xr[t];
  h0[(size_t)row * 128 + t]      = xv;
  h0[(size_t)row * 128 + 64 + t] = m;
  h0bT[(size_t)t * N_NODES + row]        = f2bf(xv);
  h0bT[(size_t)(64 + t) * N_NODES + row] = f2bf(m);
}

// ---------------------------------------------------------------------------
// C(N x F) = A(N x N, fp32 0/1 adjacency) * B(N x F bf16, given TRANSPOSED as
// BT[F][N]).  256 threads (8 waves); block tile 128 x F; K-tile 32.
//  - double-buffered LDS, one barrier per K-step
//  - A tile: coalesced float4 global loads -> regs -> v_perm_b32 pack -> LDS
//    (truncation is exact: adjacency values are only 0.0 / 1.0)
//  - B tile: CDNA5 async global->LDS b64 copies (ASYNCcnt), no VGPR staging
//  - rows padded to 18 dwords: 8B-aligned stores, conflict-free frag reads
// ---------------------------------------------------------------------------
template <int F>
__global__ __launch_bounds__(256) void agg_gemm(
    const float* __restrict__ A, const unsigned short* __restrict__ BT,
    float* __restrict__ C) {
  constexpr int NT   = F / 16;
  constexpr int KT   = N_NODES / 32;
  __shared__ unsigned int Ash[2][128 * 18];
  __shared__ unsigned int Bsh[2][F * 18];

  const int tid  = threadIdx.x;
  const int lane = tid & 31;
  const int wave = tid >> 5;
  const int hf   = lane >> 4;
  const int mrow = lane & 15;
  const int rb   = blockIdx.x;

  v8f acc[NT];
#pragma unroll
  for (int t = 0; t < NT; t++)
#pragma unroll
    for (int r = 0; r < 8; r++) acc[t][r] = 0.f;

  const int ar  = tid >> 1;         // A row within 128-row tile
  const int akb = (tid & 1) * 16;   // A float offset within 32-wide K tile

  auto loadA = [&](int kt, float4* r) {
    const float4* a4 =
        (const float4*)(A + (size_t)(rb * 128 + ar) * ADJ_LD + kt * 32 + akb);
#pragma unroll
    for (int j = 0; j < 4; j++) r[j] = a4[j];
  };
  auto storeA = [&](int b, const float4* r) {
#pragma unroll
    for (int j = 0; j < 4; j++) {
      Ash[b][ar * 18 + (akb >> 1) + j * 2 + 0] = pack2bf_trunc(r[j].x, r[j].y);
      Ash[b][ar * 18 + (akb >> 1) + j * 2 + 1] = pack2bf_trunc(r[j].z, r[j].w);
    }
  };
  // B tile: F rows x 64B; 8-byte chunks, F*8 chunks total, F/32 per thread
  auto asyncB = [&](int kt, int b) {
#pragma unroll
    for (int j = 0; j < F / 32; j++) {
      int c  = tid + j * 256;
      int n  = c >> 3;
      int kb = (c & 7) * 8;
      const char* src = (const char*)BT + (size_t)n * (N_NODES * 2) +
                        (size_t)kt * 64 + kb;
      char* dst = (char*)(&Bsh[b][0]) + n * 72 + kb;
      async_copy_b64(src, dst);
    }
  };

  // prologue: stage tile 0
  float4 areg[4];
  loadA(0, areg);
  asyncB(0, 0);
  storeA(0, areg);
  wait_async0();
  __syncthreads();

  int buf = 0;
  for (int kt = 0; kt < KT; kt++) {
    const bool more = (kt + 1) < KT;
    if (more) {            // issue next tile's traffic before computing
      loadA(kt + 1, areg);
      asyncB(kt + 1, buf ^ 1);
    }

    union { unsigned int u[8]; v16bf v; } afr, bfr;
    const int arow = wave * 16 + mrow;
#pragma unroll
    for (int v = 0; v < 8; v++) {
      // A 16x32 bf16 layout: vgpr v holds K pair (v>>2)*16 + hf*8 + (v&3)*2
      afr.u[v] = Ash[buf][arow * 18 + (v >> 2) * 8 + hf * 4 + (v & 3)];
    }
#pragma unroll
    for (int t = 0; t < NT; t++) {
      int ncol = t * 16 + mrow;
#pragma unroll
      for (int v = 0; v < 8; v++)   // B 32x16: vgpr v holds K pair (v>>2)*16 + hf*8 + (v&3)*2
        bfr.u[v] = Bsh[buf][ncol * 18 + (v >> 2) * 8 + hf * 4 + (v & 3)];
      acc[t] = __builtin_amdgcn_wmma_f32_16x16x32_bf16(
          false, afr.v, false, bfr.v, (short)0, acc[t], false, false);
    }

    if (more) {
      storeA(buf ^ 1, areg);   // waits LOADcnt as needed
      wait_async0();           // this wave's async B copies landed
    }
    __syncthreads();
    buf ^= 1;
  }

  // epilogue: C/D layout -> (M = vgpr + hf*8, N = mrow)
#pragma unroll
  for (int t = 0; t < NT; t++) {
    int col = t * 16 + mrow;
#pragma unroll
    for (int r = 0; r < 8; r++) {
      int row = rb * 128 + wave * 16 + hf * 8 + r;
      C[(size_t)row * F + col] = acc[t][r];
    }
  }
}

// ---------------------------------------------------------------------------
// out[row] = relu?( [h | agg_sel] @ W_sel + b_sel ), branch picked by mask
// (mask in {0,1} so branch select == m*hn + (1-m)*he exactly).
// Also emits the TRANSPOSED bf16 copy for the next layer's GEMM (if OUTBT).
// ---------------------------------------------------------------------------
__global__ __launch_bounds__(256) void combine_kernel(
    const float* __restrict__ H, const float* __restrict__ AGN,
    const float* __restrict__ AGE,
    const float* __restrict__ WN, const float* __restrict__ BN,
    const float* __restrict__ WE, const float* __restrict__ BE,
    const int* __restrict__ mask,
    float* __restrict__ OUT, unsigned short* __restrict__ OUTBT,
    int Fi, int Fo, int act) {
  int gid = blockIdx.x * 256 + threadIdx.x;
  int o   = gid % Fo;
  int row = gid / Fo;
  if (row >= N_NODES) return;

  int m = mask[row];
  const float* W   = m ? WN : WE;
  const float* agg = m ? AGN : AGE;
  float s = m ? BN[o] : BE[o];
  const float* hr = H   + (size_t)row * Fi;
  const float* gr = agg + (size_t)row * Fi;
  for (int k = 0; k < Fi; k++) s = fmaf(hr[k], W[k * Fo + o], s);
  for (int k = 0; k < Fi; k++) s = fmaf(gr[k], W[(Fi + k) * Fo + o], s);
  if (act) s = s > 0.f ? s : 0.f;
  OUT[(size_t)row * Fo + o] = s;
  if (OUTBT) OUTBT[(size_t)o * N_NODES + row] = f2bf(s);
}

__global__ __launch_bounds__(256) void logsoftmax_kernel(
    const float* __restrict__ L, float* __restrict__ out) {
  int i = blockIdx.x * 256 + threadIdx.x;
  if (i >= N_NODES) return;
  float a = L[2 * i], b = L[2 * i + 1];
  float mx  = fmaxf(a, b);
  float lse = mx + logf(expf(a - mx) + expf(b - mx));
  out[2 * i]     = a - lse;
  out[2 * i + 1] = b - lse;
}

// ---------------------------------------------------------------------------
extern "C" void kernel_launch(void* const* d_in, const int* in_sizes, int n_in,
                              void* d_out, int out_size, void* d_ws, size_t ws_size,
                              hipStream_t stream) {
  const float* x        = (const float*)d_in[0];
  const float* node_adj = (const float*)d_in[1];
  const float* edge_adj = (const float*)d_in[2];
  const int*   node_mask= (const int*)  d_in[3];
  const float* pn_w1 = (const float*)d_in[4];
  const float* pn_b1 = (const float*)d_in[5];
  const float* pn_w2 = (const float*)d_in[6];
  const float* pn_b2 = (const float*)d_in[7];
  const float* pn_w3 = (const float*)d_in[8];
  const float* pn_b3 = (const float*)d_in[9];
  const float* gc1_wn = (const float*)d_in[10];
  const float* gc1_bn = (const float*)d_in[11];
  const float* gc1_we = (const float*)d_in[12];
  const float* gc1_be = (const float*)d_in[13];
  const float* gc2_wn = (const float*)d_in[14];
  const float* gc2_bn = (const float*)d_in[15];
  const float* gc2_we = (const float*)d_in[16];
  const float* gc2_be = (const float*)d_in[17];
  const float* gc3_wn = (const float*)d_in[18];
  const float* gc3_bn = (const float*)d_in[19];
  const float* gc3_we = (const float*)d_in[20];
  const float* gc3_be = (const float*)d_in[21];

  char* w = (char*)d_ws;
  float*          hA    = (float*)         (w + (0u  << 20));  // 4 MB (N x 128 f32)
  unsigned short* hAbT  = (unsigned short*)(w + (4u  << 20));  // 2 MB (128 x N bf16, transposed)
  float*          hB    = (float*)         (w + (6u  << 20));  // 4 MB
  unsigned short* hBbT  = (unsigned short*)(w + (10u << 20));  // 2 MB
  float*          aggn  = (float*)         (w + (12u << 20));  // 4 MB
  float*          agge  = (float*)         (w + (16u << 20));  // 4 MB
  float*          lgts  = (float*)         (w + (20u << 20));  // 64 KB

  pointnet_kernel<<<N_NODES, 64, 0, stream>>>(x, pn_w1, pn_b1, pn_w2, pn_b2,
                                              pn_w3, pn_b3, hA, hAbT);

  // layer 1 (Fi=128 -> Fo=64, relu)
  agg_gemm<128><<<N_NODES / 128, 256, 0, stream>>>(node_adj, hAbT, aggn);
  agg_gemm<128><<<N_NODES / 128, 256, 0, stream>>>(edge_adj, hAbT, agge);
  combine_kernel<<<(N_NODES * 64) / 256, 256, 0, stream>>>(
      hA, aggn, agge, gc1_wn, gc1_bn, gc1_we, gc1_be, node_mask, hB, hBbT,
      128, 64, 1);

  // layer 2 (Fi=64 -> Fo=64, relu)
  agg_gemm<64><<<N_NODES / 128, 256, 0, stream>>>(node_adj, hBbT, aggn);
  agg_gemm<64><<<N_NODES / 128, 256, 0, stream>>>(edge_adj, hBbT, agge);
  combine_kernel<<<(N_NODES * 64) / 256, 256, 0, stream>>>(
      hB, aggn, agge, gc2_wn, gc2_bn, gc2_we, gc2_be, node_mask, hA, hAbT,
      64, 64, 1);

  // layer 3 (Fi=64 -> Fo=2, no act)
  agg_gemm<64><<<N_NODES / 128, 256, 0, stream>>>(node_adj, hAbT, aggn);
  agg_gemm<64><<<N_NODES / 128, 256, 0, stream>>>(edge_adj, hAbT, agge);
  combine_kernel<<<(N_NODES * 2) / 256, 256, 0, stream>>>(
      hA, aggn, agge, gc3_wn, gc3_bn, gc3_we, gc3_be, node_mask, lgts,
      (unsigned short*)nullptr, 64, 2, 0);

  logsoftmax_kernel<<<N_NODES / 256, 256, 0, stream>>>(lgts, (float*)d_out);
}
